// CoefficientLayer_48601849921914
// MI455X (gfx1250) — matrix operand order repaired and checked by reading, not compiled, not run on hardware
//
#include <hip/hip_runtime.h>
#include <hip/hip_bf16.h>

// ---------------------------------------------------------------------------
// CoefficientLayer: species-routed 4-layer MLP (384 -> h -> h -> 96 -> 1)
// Strategy: compact atoms by species, then 16-atom-tile WMMA f16 GEMMs.
// Memory-bound on the 100.7 MB AEV read -> vectorized b128 gather; weights
// (~0.66 MB f16) stay L2-resident and feed global_load_b128 B fragments.
// ---------------------------------------------------------------------------

typedef __attribute__((ext_vector_type(16))) _Float16 v16h;
typedef __attribute__((ext_vector_type(8)))  _Float16 v8h;
typedef __attribute__((ext_vector_type(4)))  _Float16 v4h;
typedef __attribute__((ext_vector_type(8)))  float    v8f;
typedef __attribute__((ext_vector_type(4)))  float    v4f;

// Per-species dims (H, C, N, O), K dims padded to multiples of 32 at runtime.
__constant__ int DIMS_D[4][5] = {
    {384, 160, 128, 96, 1},
    {384, 144, 112, 96, 1},
    {384, 128, 112, 96, 1},
    {384, 128, 112, 96, 1}};

// f16 element offsets of transposed, K-padded weights in the workspace.
__constant__ int WOFF_D[4][4] = {
    {0,      61440,  81920,  94208},
    {94304,  149600, 167520, 179808},
    {179904, 229056, 243392, 255680},
    {255776, 304928, 319264, 331552}};

static const int DIMS_H[4][5] = {
    {384, 160, 128, 96, 1},
    {384, 144, 112, 96, 1},
    {384, 128, 112, 96, 1},
    {384, 128, 112, 96, 1}};
static const int WOFF_H[4][4] = {
    {0,      61440,  81920,  94208},
    {94304,  149600, 167520, 179808},
    {179904, 229056, 243392, 255680},
    {255776, 304928, 319264, 331552}};

struct BiasPtrs { const float* p[4][4]; };

// --------------------------- prep kernels ----------------------------------

// Convert fp32 weight [din][dout] (row-major, (k,n) at k*dout+n) into f16
// transposed [dout][ldk] with zero K-padding.
__global__ void convert_weights(const float* __restrict__ w,
                                _Float16* __restrict__ wt,
                                int din, int dout, int ldk) {
    int tid = blockIdx.x * blockDim.x + threadIdx.x;
    int total = dout * ldk;
    if (tid >= total) return;
    int n = tid / ldk, k = tid % ldk;
    wt[tid] = (k < din) ? (_Float16)w[k * dout + n] : (_Float16)0.0f;
}

__global__ void init_counters(int* cnt) {
    if (threadIdx.x < 4) cnt[threadIdx.x] = 0;
}

// Compact atom indices into per-species lists; zero output for padding atoms.
__global__ void partition_atoms(const int* __restrict__ species,
                                int* __restrict__ cnt,
                                int* __restrict__ lists,
                                float* __restrict__ out, int natoms) {
    int gid = blockIdx.x * blockDim.x + threadIdx.x;
    if (gid >= natoms) return;
    int s = species[gid];
    if (s >= 0 && s < 4) {
        int pos = atomicAdd(&cnt[s], 1);
        lists[s * natoms + pos] = gid;
    } else {
        out[gid] = 0.0f;  // padding atoms stay 0
    }
}

// --------------------------- WMMA fragment loads ---------------------------

// A-matrix 16x32 f16 layout (ISA 7.12.2): lane L holds row M=L%16;
// lanes 0-15: K = kk*32 + {0..7, 16..23}; lanes 16-31: +8.
__device__ __forceinline__ v16h load_a_frag(const _Float16* X, int ld, int kk,
                                            int lane) {
    int row = lane & 15;
    int kbase = kk * 32 + ((lane & 16) ? 8 : 0);
    const _Float16* p = X + row * ld + kbase;
    v8h lo = *(const v8h*)p;         // K = kbase .. kbase+7
    v8h hi = *(const v8h*)(p + 16);  // K = kbase+16 .. kbase+23
    v16h a;
#pragma unroll
    for (int i = 0; i < 8; ++i) { a[i] = lo[i]; a[i + 8] = hi[i]; }
    return a;
}

// B-matrix 32x16 f16: lane holds column N=lane%16; lanes 0-15: K=kk*32+{0..7,16..23},
// lanes 16-31: +8.  Weights stored transposed [N][K] -> contiguous.
__device__ __forceinline__ v16h load_b_frag(const _Float16* __restrict__ Wt,
                                            int ld, int nt, int kk, int lane) {
    int n = nt * 16 + (lane & 15);
    int kbase = kk * 32 + ((lane & 16) ? 8 : 0);
    const _Float16* q = Wt + n * ld + kbase;
    v8h lo = *(const v8h*)q;
    v8h hi = *(const v8h*)(q + 16);
    v16h b;
#pragma unroll
    for (int i = 0; i < 8; ++i) { b[i] = lo[i]; b[i + 8] = hi[i]; }
    return b;
}

// --------------------------- main MLP kernel -------------------------------

__global__ void __launch_bounds__(128)
mlp_wmma_kernel(const float* __restrict__ aev,
                const int* __restrict__ lists,
                const int* __restrict__ cnt,
                const _Float16* __restrict__ wbase,
                BiasPtrs bp,
                const float* __restrict__ sb0, const float* __restrict__ sb1,
                float* __restrict__ out, int natoms) {
    __shared__ __align__(16) _Float16 xbuf[16 * 384];
    __shared__ __align__(16) _Float16 ybuf[16 * 160];
    __shared__ int aidx[16];

    const int s = blockIdx.y;
    const int count = cnt[s];
    const int base = blockIdx.x * 16;
    if (base >= count) return;

    const int tid = threadIdx.x;
    const int lane = tid & 31;
    const int wave = tid >> 5;

    if (tid < 16) {
        int j = base + tid;
        if (j >= count) j = count - 1;  // duplicate last valid atom (unwritten)
        aidx[tid] = lists[s * natoms + j];
    }
    __syncthreads();

    // Gather 16 AEV rows (1536 contiguous bytes each) as float4 b128 loads,
    // packed-convert to f16 (v_cvt_pk_f16_f32), 8-byte LDS stores.
    // 16 rows * 96 float4 = 1536 vectors, 12 per thread.
    {
        const int nvec = 16 * (384 / 4);
        for (int i = tid; i < nvec; i += 128) {
            int m = i / 96, c4 = i % 96;
            v4f v = *(const v4f*)(aev + (size_t)aidx[m] * 384 + c4 * 4);
            v4h h;
#pragma unroll
            for (int e = 0; e < 4; ++e) h[e] = (_Float16)v[e];
            *(v4h*)(xbuf + m * 384 + c4 * 4) = h;
        }
    }
    __syncthreads();

    _Float16* bufA = xbuf;  // current layer input
    _Float16* bufB = ybuf;  // current layer output

    for (int l = 0; l < 3; ++l) {
        const int din = DIMS_D[s][l];
        const int dout = DIMS_D[s][l + 1];
        const int ldin = (din + 31) & ~31;    // padded K stride
        const int ldout = (dout + 31) & ~31;  // padded output stride
        const _Float16* Wt = wbase + WOFF_D[s][l];
        const float* bias = bp.p[s][l];
        const int ksteps = ldin >> 5;
        const int ntiles = dout >> 4;  // real 16-wide output tiles
        const int ncol = lane & 15;
        const int mbase = (lane & 16) ? 8 : 0;

        for (int nt = wave; nt < ntiles; nt += 4) {
            v8f c = {};
            for (int kk = 0; kk < ksteps; ++kk) {
                v16h a = load_a_frag(bufA, ldin, kk, lane);
                v16h b = load_b_frag(Wt, ldin, nt, kk, lane);
                c = __builtin_amdgcn_wmma_f32_16x16x32_f16(
                    false, a, false, b, (short)0, c, false, false);
            }
            const int ng = nt * 16 + ncol;
            const float bn = bias[ng];
#pragma unroll
            for (int r = 0; r < 8; ++r) {
                float v = c[r] + bn;
                // CELU(alpha=0.1): x>0 ? x : alpha*(exp(x/alpha)-1)
                v = v > 0.0f ? v : 0.1f * (__expf(v * 10.0f) - 1.0f);
                bufB[(mbase + r) * ldout + ng] = (_Float16)v;
            }
        }
        // Zero padded columns [dout, ldout) so next layer's K-pad is clean.
        if (ldout > dout) {
            const int pw = ldout - dout;
            for (int i = tid; i < 16 * pw; i += 128) {
                int m = i / pw, cpad = i % pw;
                bufB[m * ldout + dout + cpad] = (_Float16)0.0f;
            }
        }
        __syncthreads();
        _Float16* t = bufA; bufA = bufB; bufB = t;
    }

    // Final layer: 96 -> 1 dot product + shifter (bufA holds 16 x 96 f16).
    if (tid < 16) {
        int j = base + tid;
        if (j < count) {
            const _Float16* w3 = wbase + WOFF_D[s][3];
            float sum = 0.0f;
#pragma unroll
            for (int v = 0; v < 12; ++v) {  // 12 x v8h = 96 elements
                v8h xa = *(const v8h*)(bufA + tid * 96 + v * 8);
                v8h wb = *(const v8h*)(w3 + v * 8);
#pragma unroll
                for (int e = 0; e < 8; ++e) sum += (float)xa[e] * (float)wb[e];
            }
            sum += bp.p[s][3][0];
            out[aidx[tid]] = sb0[s] + sb1[s] * sum;
        }
    }
}

// --------------------------- launcher --------------------------------------

extern "C" void kernel_launch(void* const* d_in, const int* in_sizes, int n_in,
                              void* d_out, int out_size, void* d_ws,
                              size_t ws_size, hipStream_t stream) {
    const int* species = (const int*)d_in[0];
    const float* aev = (const float*)d_in[1];
    const float* sb0 = (const float*)d_in[2];
    const float* sb1 = (const float*)d_in[3];
    const int natoms = in_sizes[0];  // B*A = 65536

    // Workspace layout: [4 counters][4 * natoms index lists][f16 weights]
    int* cnt = (int*)d_ws;
    int* lists = (int*)((char*)d_ws + 16);
    _Float16* wbase =
        (_Float16*)((char*)d_ws + 16 + (size_t)4 * natoms * sizeof(int));

    // 1) Convert all 16 weight matrices to transposed, K-padded f16.
    for (int s = 0; s < 4; ++s) {
        for (int l = 0; l < 4; ++l) {
            const float* w = (const float*)d_in[4 + s * 8 + 2 * l];
            int din = DIMS_H[s][l], dout = DIMS_H[s][l + 1];
            int ldk = (l == 3) ? din : ((din + 31) & ~31);
            int total = dout * ldk;
            convert_weights<<<(total + 255) / 256, 256, 0, stream>>>(
                w, wbase + WOFF_H[s][l], din, dout, ldk);
        }
    }

    // 2) Species compaction (+ zero padding-atom outputs).
    init_counters<<<1, 32, 0, stream>>>(cnt);
    partition_atoms<<<(natoms + 255) / 256, 256, 0, stream>>>(
        species, cnt, lists, (float*)d_out, natoms);

    // 3) WMMA MLP over 16-atom tiles, one species per grid.y.
    BiasPtrs bp;
    for (int s = 0; s < 4; ++s)
        for (int l = 0; l < 4; ++l)
            bp.p[s][l] = (const float*)d_in[4 + s * 8 + 2 * l + 1];

    dim3 grid((natoms + 15) / 16, 4, 1);
    mlp_wmma_kernel<<<grid, 128, 0, stream>>>(aev, lists, cnt, wbase, bp, sb0,
                                              sb1, (float*)d_out, natoms);
}
